// InvertedResidualBlockMEInt_63367947485645
// MI455X (gfx1250) — 16-bit and 8-bit matrix operand order repaired and checked by reading, compiled, not run on hardware
//
#include <hip/hip_runtime.h>


typedef int            v8i  __attribute__((ext_vector_type(8)));
typedef float          v8f  __attribute__((ext_vector_type(8)));
typedef unsigned int   v8u  __attribute__((ext_vector_type(8)));
typedef _Float16       v16h __attribute__((ext_vector_type(16)));

#define N_SITES 100000
#define C_IN    96
#define C_HID   576
#define C_OUT   96

#define OFF_BFRAG1 0
#define OFF_W3FRAG 73728
#define OFF_X1     184320

#define NPREP1 (36 * 2 * 8 * 32)
#define NPREP2 (6 * 18 * 8 * 32)
#define NZERO  (C_HID / 2)

__global__ void __launch_bounds__(256) prep_kernel(
    const float* __restrict__ w1, const float* __restrict__ w3,
    int* __restrict__ bfrag1, unsigned* __restrict__ w3frag,
    unsigned* __restrict__ x1zero) {
  int gid = blockIdx.x * 256 + threadIdx.x;
  if (gid < NPREP1) {
    int lane = gid & 31, j = (gid >> 5) & 7, kstep = (gid >> 8) & 1, ct = gid >> 9;
    int hl = lane >> 4, col = ct * 16 + (lane & 15);
    int packed = 0;
#pragma unroll
    for (int b = 0; b < 4; ++b) {
      int K = kstep * 64 + (j >> 2) * 32 + hl * 16 + (j & 3) * 4 + b;
      int v = 0;
      if (K < C_IN) v = __float2int_rn(w1[K * C_HID + col]);
      packed |= (v & 0xFF) << (8 * b);
    }
    bfrag1[gid] = packed;
  } else if (gid < NPREP1 + NPREP2) {
    int g = gid - NPREP1;
    int ct = g / 4608, r = g % 4608;
    int ks = r >> 8, j = (r >> 5) & 7, lane = r & 31;
    int hl = lane >> 4, col = ct * 16 + (lane & 15);
    unsigned packed = 0;
#pragma unroll
    for (int h = 0; h < 2; ++h) {
      int K = ks * 32 + (j >> 2) * 16 + hl * 8 + (j & 3) * 2 + h;
      union { _Float16 f; unsigned short u; } cv;
      cv.f = (_Float16)w3[K * C_OUT + col];
      packed |= (unsigned)cv.u << (16 * h);
    }
    w3frag[g] = packed;
  } else if (gid < NPREP1 + NPREP2 + NZERO) {
    x1zero[gid - NPREP1 - NPREP2] = 0u;
  }
}

__global__ void __launch_bounds__(256) gemm1_kernel(
    const float* __restrict__ feats, const int* __restrict__ bfrag1,
    const float* __restrict__ b1, const float* __restrict__ s1,
    _Float16* __restrict__ x1) {
  __shared__ unsigned fq[16 * C_IN / 4];
  int tid = threadIdx.x;
  int row0 = blockIdx.x * 16;
  for (int i = tid; i < 16 * C_IN / 4; i += 256) {
    int k4 = i * 4;
    const float* p = &feats[(size_t)(row0 + k4 / C_IN) * C_IN + (k4 % C_IN)];
    unsigned packed = 0;
#pragma unroll
    for (int b = 0; b < 4; ++b)
      packed |= ((unsigned)__float2int_rn(p[b]) & 0xFFu) << (8 * b);
    fq[i] = packed;
  }
  __syncthreads();

  int lane = tid & 31, wave = tid >> 5;
  int hl = lane >> 4, mrow = lane & 15;

  v8i a[2];
#pragma unroll
  for (int ks = 0; ks < 2; ++ks) {
#pragma unroll
    for (int j = 0; j < 8; ++j) {
      int K = ks * 64 + (j >> 2) * 32 + hl * 16 + (j & 3) * 4;
      a[ks][j] = (K < C_IN) ? (int)fq[(mrow * C_IN + K) >> 2] : 0;
    }
  }

  for (int ct = wave; ct < C_HID / 16; ct += 8) {
    v8i acc = {};
#pragma unroll
    for (int ks = 0; ks < 2; ++ks) {
      v8i bf;
      const int* bp = bfrag1 + (ct * 2 + ks) * 256;
#pragma unroll
      for (int j = 0; j < 8; ++j) bf[j] = bp[j * 32 + lane];
      acc = __builtin_amdgcn_wmma_i32_16x16x64_iu8(true, a[ks], true, bf, acc,
                                                   false, false);
    }
    int col = ct * 16 + mrow;
    float bias = b1[col];
    float scale = __builtin_rintf(s1[col]) * (1.0f / 256.0f);
#pragma unroll
    for (int j = 0; j < 8; ++j) {
      int m = j + 8 * hl;
      float v = ((float)acc[j] + bias) * scale;
      v = fminf(fmaxf(v, 0.0f), 6.0f);
      x1[(size_t)(row0 + m) * C_HID + col] = (_Float16)v;
    }
  }
}

__global__ void __launch_bounds__(256) fused_dw_gemm3_kernel(
    const _Float16* __restrict__ x1, const int* __restrict__ nbr,
    const float* __restrict__ w2, const float* __restrict__ b2,
    const float* __restrict__ s2, const unsigned* __restrict__ w3frag,
    const float* __restrict__ b3, const float* __restrict__ s3,
    const float* __restrict__ feats, const float* __restrict__ s_main,
    const float* __restrict__ s_res, float* __restrict__ out) {
  __shared__ _Float16 x2t[16 * C_HID];
  __shared__ int nb[9 * 16];
  int tid = threadIdx.x;
  int row0 = blockIdx.x * 16;

  if (tid < 144) {
    int k = tid / 16, m = tid % 16;
    nb[tid] = nbr[(size_t)k * N_SITES + row0 + m];
  }
  __syncthreads();

  {
    int m = tid >> 4, sub = tid & 15;
    const _Float16* xrow[9];
#pragma unroll
    for (int k = 0; k < 9; ++k) {
      int idx = nb[k * 16 + m];
      if (idx < 0) idx = N_SITES;
      xrow[k] = x1 + (size_t)idx * C_HID;
    }
    for (int i = 0; i < C_HID / 32; ++i) {
      int c = 2 * sub + 32 * i;
      float a0 = 0.0f, a1 = 0.0f;
#pragma unroll
      for (int k = 0; k < 9; ++k) {
        union { unsigned u; _Float16 h[2]; } xv;
        xv.u = *(const unsigned*)(xrow[k] + c);
        const float* wp = &w2[k * C_HID + c];
        a0 += (float)xv.h[0] * wp[0];
        a1 += (float)xv.h[1] * wp[1];
      }
      float q0 = (a0 + b2[c])     * __builtin_rintf(s2[c])     * (1.0f / 256.0f);
      float q1 = (a1 + b2[c + 1]) * __builtin_rintf(s2[c + 1]) * (1.0f / 256.0f);
      q0 = fminf(fmaxf(q0, 0.0f), 6.0f);
      q1 = fminf(fmaxf(q1, 0.0f), 6.0f);
      union { unsigned u; _Float16 h[2]; } ov;
      ov.h[0] = (_Float16)q0;
      ov.h[1] = (_Float16)q1;
      *(unsigned*)&x2t[m * C_HID + c] = ov.u;
    }
  }
  __syncthreads();

  int wave = tid >> 5;
  if (wave >= 6) return;
  int lane = tid & 31, hl = lane >> 4, mrow = lane & 15;
  int ct = wave;

  v8f acc = {};
  for (int ks = 0; ks < C_HID / 32; ++ks) {
    union { v8u u; v16h h; } au, bu;
#pragma unroll
    for (int j = 0; j < 8; ++j) {
      int K = ks * 32 + (j >> 2) * 16 + hl * 8 + (j & 3) * 2;
      au.u[j] = *(const unsigned*)&x2t[mrow * C_HID + K];
    }
    const unsigned* bp = w3frag + (ct * 18 + ks) * 256;
#pragma unroll
    for (int j = 0; j < 8; ++j) bu.u[j] = bp[j * 32 + lane];
    acc = __builtin_amdgcn_wmma_f32_16x16x32_f16(false, au.h, false, bu.h,
                                                 (short)0, acc, false, false);
  }

  float sm = __builtin_rintf(s_main[0]) * 256.0f;
  float sr = __builtin_rintf(s_res[0]) * 256.0f;
  int col = ct * 16 + mrow;
  float bias = b3[col];
  float scale = __builtin_rintf(s3[col]) * (1.0f / 256.0f);
#pragma unroll
  for (int j = 0; j < 8; ++j) {
    int m = j + 8 * hl;
    size_t n = (size_t)(row0 + m);
    float v = (acc[j] + bias) * scale;
    v = fminf(fmaxf(v, -128.0f), 128.0f);
    out[n * C_OUT + col] = sm * v + sr * feats[n * C_OUT + col];
  }
}

extern "C" void kernel_launch(void* const* d_in, const int* in_sizes, int n_in,
                              void* d_out, int out_size, void* d_ws,
                              size_t ws_size, hipStream_t stream) {
  const float* feats  = (const float*)d_in[0];
  const int*   nbr    = (const int*)d_in[1];
  const float* w1     = (const float*)d_in[2];
  const float* b1     = (const float*)d_in[3];
  const float* w2     = (const float*)d_in[4];
  const float* b2     = (const float*)d_in[5];
  const float* w3     = (const float*)d_in[6];
  const float* b3     = (const float*)d_in[7];
  const float* s1     = (const float*)d_in[8];
  const float* s2     = (const float*)d_in[9];
  const float* s3     = (const float*)d_in[10];
  const float* s_main = (const float*)d_in[11];
  const float* s_res  = (const float*)d_in[12];
  float* out = (float*)d_out;

  char* ws = (char*)d_ws;
  int*       bfrag1 = (int*)(ws + OFF_BFRAG1);
  unsigned*  w3frag = (unsigned*)(ws + OFF_W3FRAG);
  _Float16*  x1     = (_Float16*)(ws + OFF_X1);
  unsigned*  x1zero = (unsigned*)(ws + OFF_X1 + (size_t)N_SITES * C_HID * 2);

  int tiles = N_SITES / 16;
  int prep_threads = NPREP1 + NPREP2 + NZERO;
  int prep_blocks = (prep_threads + 255) / 256;

  hipLaunchKernelGGL(prep_kernel, dim3(prep_blocks), dim3(256), 0, stream,
                     w1, w3, bfrag1, w3frag, x1zero);
  hipLaunchKernelGGL(gemm1_kernel, dim3(tiles), dim3(256), 0, stream,
                     feats, bfrag1, b1, s1, x1);
  hipLaunchKernelGGL(fused_dw_gemm3_kernel, dim3(tiles), dim3(256), 0, stream,
                     x1, nbr, w2, b2, s2, w3frag, b3, s3, feats, s_main, s_res,
                     out);
}
